// RewaHierarchicalAttention_90237262889106
// MI455X (gfx1250) — matrix operand order repaired and checked by reading, not compiled, not run on hardware
//
#include <hip/hip_runtime.h>
#include <hip/hip_bf16.h>

#define DEV __device__ __forceinline__

typedef __attribute__((ext_vector_type(16))) __bf16 v16bf;
typedef __attribute__((ext_vector_type(8)))  __bf16 v8bf;
typedef __attribute__((ext_vector_type(8)))  float  v8f;

static constexpr int B_ = 2, N_ = 4096, E_ = 1024, H_ = 16, D_ = 64;
static constexpr int M_ = B_ * N_;   // 8192 rows for the projections

// ---------------------------------------------------------------------------
// WMMA helpers (CDNA5 wave32, v_wmma_f32_16x16x32_bf16)
// ---------------------------------------------------------------------------
DEV v8f wmma_bf16(v16bf a, v16bf b, v8f c) {
  return __builtin_amdgcn_wmma_f32_16x16x32_bf16(false, a, false, b, (short)0, c,
                                                 false, false);
}

// A-matrix fragment, 16x32 bf16, row-major source [rows, ld].
// Lanes 0-15: row=lane,   K = {0..7, 16..23}
// Lanes 16-31: row=lane-16, K = {8..15, 24..31}
DEV v16bf frag_load_a(const __bf16* p, int ld, int row_base, int k_base, int lane) {
  const int m = lane & 15;
  const int o = (lane >> 4) << 3;            // 0 or 8
  const __bf16* r = p + (size_t)(row_base + m) * ld + k_base;
  v8bf lo = *(const v8bf*)(r + o);
  v8bf hi = *(const v8bf*)(r + 16 + o);
  v16bf f;
#pragma unroll
  for (int e = 0; e < 8; ++e) { f[e] = lo[e]; f[8 + e] = hi[e]; }
  return f;
}

// A fragment with per-row gather through an index array (sorted order).
DEV v16bf frag_load_a_gather(const __bf16* base, const int* ridx, int k_base, int lane) {
  const int m = lane & 15;
  const int o = (lane >> 4) << 3;
  const __bf16* r = base + (size_t)ridx[m] * D_ + k_base;
  v8bf lo = *(const v8bf*)(r + o);
  v8bf hi = *(const v8bf*)(r + 16 + o);
  v16bf f;
#pragma unroll
  for (int e = 0; e < 8; ++e) { f[e] = lo[e]; f[8 + e] = hi[e]; }
  return f;
}

// B-matrix fragment, 32x16 bf16, from B^T storage [cols(N), ld=K] row-major.
// Lanes 0-15: col=lane, K={0..7,16..23} ; lanes 16-31: col=lane-16, K={8..15,24..31}
DEV v16bf frag_load_b(const __bf16* p, int ld, int col_base, int k_base, int lane) {
  const int n = lane & 15;
  const int o = (lane >> 4) << 3;            // 0 or 8
  const __bf16* r = p + (size_t)(col_base + n) * ld + k_base;
  v8bf lo = *(const v8bf*)(r + o);
  v8bf hi = *(const v8bf*)(r + 16 + o);
  v16bf f;
#pragma unroll
  for (int e = 0; e < 8; ++e) { f[e] = lo[e]; f[8 + e] = hi[e]; }
  return f;
}

// ---------------------------------------------------------------------------
// Elementwise / prep kernels
// ---------------------------------------------------------------------------
__global__ void f32_to_bf16_kernel(const float* __restrict__ s,
                                   __bf16* __restrict__ d, int n) {
  int i = blockIdx.x * blockDim.x + threadIdx.x;
  if (i < n) d[i] = (__bf16)s[i];
}

// W fp32 [K, Nc] row-major  ->  Wt bf16 [Nc, K]
__global__ void transpose_bf16_kernel(const float* __restrict__ W,
                                      __bf16* __restrict__ Wt, int K, int Nc) {
  int i = blockIdx.x * blockDim.x + threadIdx.x;
  if (i < K * Nc) {
    int k = i % K, n = i / K;
    Wt[i] = (__bf16)W[(size_t)k * Nc + n];
  }
}

// Stable counting sort per (level, batch): matches jnp.argsort(stable).
__global__ void bucket_sort_kernel(const int* __restrict__ wb0,
                                   const int* __restrict__ wb1,
                                   const int* __restrict__ wb2,
                                   int* __restrict__ sorted) {
  if (threadIdx.x != 0) return;
  const int job = blockIdx.x;                 // lvl*B + b, 6 jobs
  const int lvl = job >> 1, b = job & 1;
  const int* wb = (lvl == 0) ? wb0 : (lvl == 1) ? wb1 : wb2;
  const int nb  = (lvl == 0) ? (N_ / 512) : (lvl == 1) ? (N_ / 128) : (N_ / 32);
  const int* src = wb + b * N_;
  int* dst = sorted + job * N_;
  int cnt[128];
  for (int i = 0; i < nb; ++i) cnt[i] = 0;
  for (int i = 0; i < N_; ++i) cnt[src[i]]++;
  int run = 0;
  for (int i = 0; i < nb; ++i) { int c = cnt[i]; cnt[i] = run; run += c; }
  for (int i = 0; i < N_; ++i) { int bu = src[i]; dst[cnt[bu]++] = i; }
}

// attn fp32 [B*H, N, D] (already carries the 1/3 average) -> ctx bf16 [B*N, E]
__global__ void build_ctx_kernel(const float* __restrict__ att,
                                 __bf16* __restrict__ ctx) {
  int i = blockIdx.x * blockDim.x + threadIdx.x;
  if (i < M_ * E_) {
    int row = i >> 10, col = i & (E_ - 1);
    int b = row >> 12, n = row & (N_ - 1);
    int h = col >> 6,  d = col & (D_ - 1);
    ctx[i] = (__bf16)att[((size_t)((b * H_ + h) * N_) + n) * D_ + d];
  }
}

// ---------------------------------------------------------------------------
// WMMA GEMM:  C[M,Nc] = A[M,K]_bf16 * Bt[Nc,K]_bf16^T + bias
// MODE 0: fp32 store to Cf [M,Nc]   (final projection -> d_out)
// MODE 1: bf16 store head-split to Cbf [(b*H+h), n, d]  (Q/K/V)
// Block: 256 threads / 8 waves, tile 128(M) x 64(N), K-step 32.
// ---------------------------------------------------------------------------
template <int MODE>
__global__ __launch_bounds__(256)
void gemm_bf16_kernel(const __bf16* __restrict__ A, const __bf16* __restrict__ Bt,
                      const float* __restrict__ bias, float* __restrict__ Cf,
                      __bf16* __restrict__ Cbf, int K, int Nc) {
  const int lane = threadIdx.x & 31;
  const int wave = threadIdx.x >> 5;
  const int row0 = blockIdx.x * 128 + wave * 16;
  const int col0 = blockIdx.y * 64;

  v8f acc[4] = {};
  for (int k = 0; k < K; k += 32) {
    v16bf af = frag_load_a(A, K, row0, k, lane);
#pragma unroll
    for (int nf = 0; nf < 4; ++nf) {
      v16bf bf = frag_load_b(Bt, K, col0 + nf * 16, k, lane);
      acc[nf] = wmma_bf16(af, bf, acc[nf]);
    }
  }

  const int half = (lane >> 4) << 3;
#pragma unroll
  for (int nf = 0; nf < 4; ++nf) {
    const int col = col0 + nf * 16 + (lane & 15);
    const float bv = bias[col];
#pragma unroll
    for (int v = 0; v < 8; ++v) {
      const int row = row0 + v + half;
      const float val = acc[nf][v] + bv;
      if (MODE == 0) {
        Cf[(size_t)row * Nc + col] = val;
      } else {
        const int b = row >> 12, n = row & (N_ - 1);
        const int h = col >> 6,  d = col & (D_ - 1);
        Cbf[((size_t)((b * H_ + h) * N_) + n) * D_ + d] = (__bf16)val;
      }
    }
  }
}

// ---------------------------------------------------------------------------
// Flash-style chunked attention for one level.
// grid: (N/MB, B*H); MB = WAVES*16 queries per block; keys = 2*CHUNK streamed
// in 64-key tiles. Chunk 0 stages zeros for the "previous chunk" half,
// reproducing the reference's zero-score softmax terms.
// ---------------------------------------------------------------------------
template <int CHUNK, int WAVES>
__global__ __launch_bounds__(WAVES * 32)
void attn_level_kernel(const __bf16* __restrict__ qh, const __bf16* __restrict__ kh,
                       const __bf16* __restrict__ vh, const int* __restrict__ sortedAll,
                       float* __restrict__ outacc, int lvl, int accumulate) {
  constexpr int MB  = WAVES * 16;
  constexpr int NT  = WAVES * 32;
  constexpr int NKT = (2 * CHUNK) / 64;

  __shared__ __attribute__((aligned(64))) __bf16 Ks[64][64];          // [key][dim]
  __shared__ __attribute__((aligned(64))) __bf16 Vt[64][64];          // [dim][key]
  __shared__ __attribute__((aligned(64))) __bf16 Pl[WAVES][16][64];   // P transpose buf

  const int tid  = threadIdx.x;
  const int lane = tid & 31;
  const int wave = tid >> 5;
  const int bh   = blockIdx.y;
  const int b    = bh >> 4;
  const int q0   = blockIdx.x * MB;
  const int chunk = q0 / CHUNK;

  const int* sidx = sortedAll + (lvl * B_ + b) * N_;
  const __bf16* qb = qh + (size_t)bh * N_ * D_;
  const __bf16* kb = kh + (size_t)bh * N_ * D_;
  const __bf16* vb = vh + (size_t)bh * N_ * D_;

  const int qrow0 = q0 + wave * 16;
  const v16bf Qf0 = frag_load_a_gather(qb, sidx + qrow0, 0, lane);
  const v16bf Qf1 = frag_load_a_gather(qb, sidx + qrow0, 32, lane);

  v8f O[4] = {};
  float mrun[8], lrun[8];
#pragma unroll
  for (int v = 0; v < 8; ++v) { mrun[v] = -1e30f; lrun[v] = 0.f; }

  const float sc = 0.125f;                       // 1/sqrt(D)
  const int kfirst = (chunk - 1) * CHUNK;        // prev-chunk start (may be <0)

  for (int kt = 0; kt < NKT; ++kt) {
    const int tk0 = kfirst + kt * 64;

    // ---- stage 64 gathered K rows + transposed V rows into LDS ----
    for (int e = tid; e < 64 * 8; e += NT) {
      const int r = e >> 3, c8 = (e & 7) << 3;
      const int kpos = tk0 + r;
      v8bf kv, vv;
      if (kpos < 0) {
#pragma unroll
        for (int j = 0; j < 8; ++j) { kv[j] = (__bf16)0.f; vv[j] = (__bf16)0.f; }
      } else {
        const size_t srow = (size_t)sidx[kpos] * D_;
        kv = *(const v8bf*)(kb + srow + c8);
        vv = *(const v8bf*)(vb + srow + c8);
      }
      *(v8bf*)&Ks[r][c8] = kv;
#pragma unroll
      for (int j = 0; j < 8; ++j) Vt[c8 + j][r] = vv[j];
    }
    __syncthreads();

    // ---- S = Q * K^T for this 64-key tile (4 N-subtiles, K-dim 64) ----
    v8f S[4];
#pragma unroll
    for (int nf = 0; nf < 4; ++nf) {
      v16bf b0 = frag_load_b(&Ks[0][0], 64, nf * 16, 0, lane);
      v16bf b1 = frag_load_b(&Ks[0][0], 64, nf * 16, 32, lane);
      v8f z = {};
      z = wmma_bf16(Qf0, b0, z);
      S[nf] = wmma_bf16(Qf1, b1, z);
    }
#pragma unroll
    for (int nf = 0; nf < 4; ++nf)
#pragma unroll
      for (int v = 0; v < 8; ++v) S[nf][v] *= sc;

    // ---- online softmax update (row reductions across 16-lane halves) ----
    float corr[8];
#pragma unroll
    for (int v = 0; v < 8; ++v) {
      float rm = fmaxf(fmaxf(S[0][v], S[1][v]), fmaxf(S[2][v], S[3][v]));
#pragma unroll
      for (int msk = 1; msk < 16; msk <<= 1) rm = fmaxf(rm, __shfl_xor(rm, msk, 32));
      const float nm = fmaxf(mrun[v], rm);
      corr[v] = __expf(mrun[v] - nm);
      mrun[v] = nm;
      float rs = 0.f;
#pragma unroll
      for (int nf = 0; nf < 4; ++nf) {
        const float p = __expf(S[nf][v] - nm);
        S[nf][v] = p;
        rs += p;
      }
#pragma unroll
      for (int msk = 1; msk < 16; msk <<= 1) rs += __shfl_xor(rs, msk, 32);
      lrun[v] = lrun[v] * corr[v] + rs;
    }

    // ---- rescale O, write P (bf16) to wave-private LDS for A-layout ----
#pragma unroll
    for (int nf = 0; nf < 4; ++nf)
#pragma unroll
      for (int v = 0; v < 8; ++v) {
        O[nf][v] *= corr[v];
        const int row = v + ((lane >> 4) << 3);
        Pl[wave][row][nf * 16 + (lane & 15)] = (__bf16)S[nf][v];
      }
    asm volatile("s_wait_dscnt 0" ::: "memory");

    const v16bf Pf0 = frag_load_a(&Pl[wave][0][0], 64, 0, 0, lane);
    const v16bf Pf1 = frag_load_a(&Pl[wave][0][0], 64, 0, 32, lane);

    // ---- O += P * V  (Vt is the B^T layout) ----
#pragma unroll
    for (int nf = 0; nf < 4; ++nf) {
      v16bf b0 = frag_load_b(&Vt[0][0], 64, nf * 16, 0, lane);
      v16bf b1 = frag_load_b(&Vt[0][0], 64, nf * 16, 32, lane);
      O[nf] = wmma_bf16(Pf0, b0, O[nf]);
      O[nf] = wmma_bf16(Pf1, b1, O[nf]);
    }
    __syncthreads();
  }

  // ---- normalize, 1/3 level average, scatter-add through sorted index ----
  const int half = (lane >> 4) << 3;
#pragma unroll
  for (int v = 0; v < 8; ++v) {
    const float s = (1.0f / 3.0f) / lrun[v];
    const int row = qrow0 + v + half;
    const size_t drow = ((size_t)bh * N_ + sidx[row]) * D_;
#pragma unroll
    for (int nf = 0; nf < 4; ++nf) {
      const size_t off = drow + nf * 16 + (lane & 15);
      float val = O[nf][v] * s;
      if (accumulate) val += outacc[off];
      outacc[off] = val;
    }
  }
}

// ---------------------------------------------------------------------------
// Host orchestration
// ---------------------------------------------------------------------------
extern "C" void kernel_launch(void* const* d_in, const int* in_sizes, int n_in,
                              void* d_out, int out_size, void* d_ws, size_t ws_size,
                              hipStream_t stream) {
  (void)in_sizes; (void)n_in; (void)out_size; (void)ws_size;
  const float* x  = (const float*)d_in[0];
  const float* Wq = (const float*)d_in[1]; const float* bq = (const float*)d_in[2];
  const float* Wk = (const float*)d_in[3]; const float* bk = (const float*)d_in[4];
  const float* Wv = (const float*)d_in[5]; const float* bv = (const float*)d_in[6];
  const float* Wo = (const float*)d_in[7]; const float* bo = (const float*)d_in[8];
  const int* wbc = (const int*)d_in[9];
  const int* wbm = (const int*)d_in[10];
  const int* wbf = (const int*)d_in[11];

  char* ws = (char*)d_ws;
  size_t off = 0;
  auto alloc = [&](size_t bytes) -> void* {
    void* p = ws + off;
    off = (off + bytes + 255) & ~(size_t)255;
    return p;
  };
  __bf16* x_bf = (__bf16*)alloc((size_t)M_ * E_ * 2);
  __bf16* Wtq  = (__bf16*)alloc((size_t)E_ * E_ * 2);
  __bf16* Wtk  = (__bf16*)alloc((size_t)E_ * E_ * 2);
  __bf16* Wtv  = (__bf16*)alloc((size_t)E_ * E_ * 2);
  __bf16* Wto  = (__bf16*)alloc((size_t)E_ * E_ * 2);
  __bf16* qh   = (__bf16*)alloc((size_t)B_ * H_ * N_ * D_ * 2);
  __bf16* kh   = (__bf16*)alloc((size_t)B_ * H_ * N_ * D_ * 2);
  __bf16* vh   = (__bf16*)alloc((size_t)B_ * H_ * N_ * D_ * 2);
  float*  attn = (float*)alloc((size_t)B_ * H_ * N_ * D_ * 4);
  __bf16* ctx  = (__bf16*)alloc((size_t)M_ * E_ * 2);
  int* sorted  = (int*)alloc((size_t)3 * B_ * N_ * 4);

  const int nx = M_ * E_;
  f32_to_bf16_kernel<<<(nx + 255) / 256, 256, 0, stream>>>(x, x_bf, nx);
  const int nw = E_ * E_;
  transpose_bf16_kernel<<<(nw + 255) / 256, 256, 0, stream>>>(Wq, Wtq, E_, E_);
  transpose_bf16_kernel<<<(nw + 255) / 256, 256, 0, stream>>>(Wk, Wtk, E_, E_);
  transpose_bf16_kernel<<<(nw + 255) / 256, 256, 0, stream>>>(Wv, Wtv, E_, E_);
  transpose_bf16_kernel<<<(nw + 255) / 256, 256, 0, stream>>>(Wo, Wto, E_, E_);

  dim3 ggrid(M_ / 128, E_ / 64);
  gemm_bf16_kernel<1><<<ggrid, 256, 0, stream>>>(x_bf, Wtq, bq, nullptr, qh, E_, E_);
  gemm_bf16_kernel<1><<<ggrid, 256, 0, stream>>>(x_bf, Wtk, bk, nullptr, kh, E_, E_);
  gemm_bf16_kernel<1><<<ggrid, 256, 0, stream>>>(x_bf, Wtv, bv, nullptr, vh, E_, E_);

  bucket_sort_kernel<<<6, 32, 0, stream>>>(wbc, wbm, wbf, sorted);

  attn_level_kernel<512, 8><<<dim3(N_ / 128, B_ * H_), 256, 0, stream>>>(
      qh, kh, vh, sorted, attn, 0, 0);
  attn_level_kernel<128, 8><<<dim3(N_ / 128, B_ * H_), 256, 0, stream>>>(
      qh, kh, vh, sorted, attn, 1, 1);
  attn_level_kernel<32, 2><<<dim3(N_ / 32, B_ * H_), 64, 0, stream>>>(
      qh, kh, vh, sorted, attn, 2, 1);

  build_ctx_kernel<<<(nx + 255) / 256, 256, 0, stream>>>(attn, ctx);

  gemm_bf16_kernel<0><<<ggrid, 256, 0, stream>>>(ctx, Wto, bo, (float*)d_out,
                                                 nullptr, E_, E_);
}
